// PointerNetDecoder_83897891160933
// MI455X (gfx1250) — matrix operand order repaired and checked by reading, not compiled, not run on hardware
//
#include <hip/hip_runtime.h>
#include <math.h>

// Problem constants (match reference)
#define NB   512          // batch
#define SEQ  128          // steps / sequence positions
#define HID  512          // hidden
#define INDIM 2
#define G4   (4*HID)      // 2048 gates
#define INF_ 1.0e7f

// ---------- types ----------
typedef __attribute__((ext_vector_type(16))) __bf16          bf16x16;
typedef __attribute__((ext_vector_type(16))) unsigned short  u16x16;
typedef __attribute__((ext_vector_type(8)))  unsigned short  u16x8;
typedef __attribute__((ext_vector_type(8)))  float           f32x8;

__device__ __forceinline__ unsigned short f2bf(float f) {
  unsigned int u = __float_as_uint(f);
  unsigned int r = (u + 0x7FFFu + ((u >> 16) & 1u)) >> 16;  // RNE
  return (unsigned short)r;
}
__device__ __forceinline__ float bf2f(unsigned short h) {
  return __uint_as_float(((unsigned int)h) << 16);
}

// ---------- WMMA GEMM: C[M,N] = A[M,K](f32, converted) * Bcm[N,K](bf16, col-major) ----------
// MODE 0: store f32   MODE 1: store bf16   MODE 2: gates epilogue (+biases +prev@W_ih.T)
#define BM 128
#define BN 64
#define BK 32

template<int MODE>
__global__ __launch_bounds__(256)
void gemm_k(const float* __restrict__ A, const unsigned short* __restrict__ Bcm,
            float* __restrict__ C, unsigned short* __restrict__ Cb,
            int M, int N, int K,
            const float* __restrict__ bih, const float* __restrict__ bhh,
            const float* __restrict__ Wih, const float* __restrict__ prev)
{
  // padded row strides (40 ushorts = 80 B) keep every b128 fragment access 16B-aligned
  __shared__ unsigned short As[BM][BK + 8];
  __shared__ unsigned short Bs[BN][BK + 8];   // n-major: lane fragments are K-contiguous

  const int tid  = threadIdx.x;
  const int wid  = tid >> 5;
  const int lane = tid & 31;
  const int half = lane >> 4;
  const int l16  = lane & 15;

  const int bm0 = blockIdx.y * BM;
  const int bn0 = blockIdx.x * BN;

  f32x8 acc[4];
#pragma unroll
  for (int nt = 0; nt < 4; ++nt)
#pragma unroll
    for (int i = 0; i < 8; ++i) acc[nt][i] = 0.0f;

  // register double-buffer for the global->LDS pipeline
  float4  aReg[4];
  ushort4 bReg[2];

  auto load_global = [&](int k0) {
#pragma unroll
    for (int t = 0; t < 4; ++t) {
      int c   = tid + t * 256;
      int row = c >> 3;                    // 8 float4 chunks per A row
      int col = (c & 7) << 2;
      aReg[t] = *(const float4*)(A + (size_t)(bm0 + row) * K + k0 + col);
    }
#pragma unroll
    for (int t = 0; t < 2; ++t) {
      int c    = tid + t * 256;            // 0..511
      int nrow = c >> 3;                   // 8 ushort4 chunks per B n-row
      int koff = (c & 7) << 2;
      bReg[t] = *(const ushort4*)(Bcm + (size_t)(bn0 + nrow) * K + k0 + koff);
    }
  };

  auto store_lds = [&]() {
#pragma unroll
    for (int t = 0; t < 4; ++t) {
      int c   = tid + t * 256;
      int row = c >> 3;
      int col = (c & 7) << 2;
      As[row][col + 0] = f2bf(aReg[t].x);
      As[row][col + 1] = f2bf(aReg[t].y);
      As[row][col + 2] = f2bf(aReg[t].z);
      As[row][col + 3] = f2bf(aReg[t].w);
    }
#pragma unroll
    for (int t = 0; t < 2; ++t) {
      int c    = tid + t * 256;
      int nrow = c >> 3;
      int koff = (c & 7) << 2;
      *(ushort4*)&Bs[nrow][koff] = bReg[t];
    }
  };

  const int NT = K / BK;
  load_global(0);
  store_lds();
  __syncthreads();

  for (int kt = 0; kt < NT; ++kt) {
    const bool more = (kt + 1 < NT);
    if (more) load_global((kt + 1) * BK);              // hide global latency behind WMMA
    if (kt + 2 < NT) {                                 // warm L2 two tiles ahead
      __builtin_prefetch(A + (size_t)(bm0 + (tid >> 1)) * K + (kt + 2) * BK, 0, 3);
      __builtin_prefetch(Bcm + (size_t)(bn0 + (tid & 63)) * K + (kt + 2) * BK, 0, 3);
    }

    // A fragment (16x32 bf16, ISA 7.12.2): slots 0..7 = K=half*8+0..7, slots 8..15 = K=16+half*8+0..7
    const unsigned short* arow = &As[wid * 16 + l16][0];
    u16x8 a0 = *(const u16x8*)(arow + half * 8);
    u16x8 a1 = *(const u16x8*)(arow + 16 + half * 8);
    u16x16 av = __builtin_shufflevector(a0, a1, 0, 1, 2, 3, 4, 5, 6, 7,
                                                8, 9, 10, 11, 12, 13, 14, 15);
    bf16x16 af = __builtin_bit_cast(bf16x16, av);

    // load ALL B fragments first, then issue the 4 WMMAs back-to-back
    bf16x16 bf[4];
#pragma unroll
    for (int nt = 0; nt < 4; ++nt) {
      const unsigned short* brow = &Bs[nt * 16 + l16][half * 8];
      u16x8 b0 = *(const u16x8*)(brow);
      u16x8 b1 = *(const u16x8*)(brow + 16);
      u16x16 bv = __builtin_shufflevector(b0, b1, 0, 1, 2, 3, 4, 5, 6, 7,
                                                  8, 9, 10, 11, 12, 13, 14, 15);
      bf[nt] = __builtin_bit_cast(bf16x16, bv);
    }
#pragma unroll
    for (int nt = 0; nt < 4; ++nt)
      acc[nt] = __builtin_amdgcn_wmma_f32_16x16x32_bf16(
          false, af, false, bf[nt], (short)0, acc[nt], false, false);

    if (more) {
      __syncthreads();        // all waves done reading current tile
      store_lds();            // commit next tile (regs -> LDS)
      __syncthreads();        // next tile visible
    }
  }

  // epilogue: C/D layout — VGPR i holds M=i (lanes0-15) / M=8+i (lanes16-31), N=lane&15
  const int m0 = bm0 + wid * 16 + half * 8;
#pragma unroll
  for (int nt = 0; nt < 4; ++nt) {
    int n = bn0 + nt * 16 + l16;
#pragma unroll
    for (int i = 0; i < 8; ++i) {
      int m = m0 + i;
      float val = acc[nt][i];
      if (MODE == 1) {
        Cb[(size_t)m * N + n] = f2bf(val);
      } else if (MODE == 2) {
        val += bih[n] + bhh[n]
             + prev[m * 2 + 0] * Wih[n * 2 + 0]
             + prev[m * 2 + 1] * Wih[n * 2 + 1];
        C[(size_t)m * N + n] = val;
      } else {
        C[(size_t)m * N + n] = val;
      }
    }
  }
}

// ---------- LSTM pointwise: gates[N,4H] (i,f,g,o) -> c,h ----------
__global__ __launch_bounds__(256)
void lstm_pointwise(const float* __restrict__ gates, float* __restrict__ c,
                    float* __restrict__ h)
{
  int idx = blockIdx.x * 256 + threadIdx.x;   // < NB*HID
  int n  = idx >> 9;
  int hh = idx & (HID - 1);
  const float* g = gates + (size_t)n * G4;
  float gi = g[hh], gf = g[hh + HID], gg = g[hh + 2 * HID], go = g[hh + 3 * HID];
  float si = 1.0f / (1.0f + __expf(-gi));
  float sf = 1.0f / (1.0f + __expf(-gf));
  float so = 1.0f / (1.0f + __expf(-go));
  float cn = sf * c[idx] + si * tanhf(gg);
  c[idx] = cn;
  h[idx] = so * tanhf(cn);
}

// ---------- attention + log-softmax + argmax + state update (1 block per batch row) ----------
__global__ __launch_bounds__(256)
void attention_step(const unsigned short* __restrict__ rW, const float* __restrict__ q,
                    const float* __restrict__ vvec, const float* __restrict__ x,
                    float* __restrict__ mask, float* __restrict__ ll,
                    float* __restrict__ prev, float* __restrict__ tour, int step)
{
  __shared__ float sc[SEQ];
  const int n    = blockIdx.x;
  const int tid  = threadIdx.x;
  const int wid  = tid >> 5;
  const int lane = tid & 31;
  const float* qn = q + (size_t)n * HID;

  for (int s = wid; s < SEQ; s += 8) {
    const unsigned short* row = rW + ((size_t)n * SEQ + s) * HID;
    if (s + 8 < SEQ)  // prefetch next row this wave will process (global_prefetch_b8)
      __builtin_prefetch(rW + ((size_t)n * SEQ + s + 8) * HID + lane * 16, 0, 3);
    float p = 0.0f;
#pragma unroll
    for (int it = 0; it < 4; ++it) {
      int hb = it * 128 + lane * 4;
      const ushort4 rv = *(const ushort4*)(row + hb);
      const float4  qv = *(const float4*)(qn + hb);
      const float4  vl = *(const float4*)(vvec + hb);
      p += vl.x * tanhf(bf2f(rv.x) + qv.x);
      p += vl.y * tanhf(bf2f(rv.y) + qv.y);
      p += vl.z * tanhf(bf2f(rv.z) + qv.z);
      p += vl.w * tanhf(bf2f(rv.w) + qv.w);
    }
#pragma unroll
    for (int off = 16; off > 0; off >>= 1) p += __shfl_xor(p, off, 32);
    if (lane == 0) sc[s] = p - INF_ * mask[(size_t)n * SEQ + s];
  }
  __syncthreads();

  if (tid == 0) {
    float m = sc[0]; int pos = 0;
    for (int s = 1; s < SEQ; ++s) if (sc[s] > m) { m = sc[s]; pos = s; }
    float sum = 0.0f;
    for (int s = 0; s < SEQ; ++s) sum += __expf(sc[s] - m);
    ll[n] += -__logf(sum);                    // logp[argmax] = -log(sum exp(sc - max))
    mask[(size_t)n * SEQ + pos] = 1.0f;
    prev[n * 2 + 0] = x[((size_t)n * SEQ + pos) * INDIM + 0];
    prev[n * 2 + 1] = x[((size_t)n * SEQ + pos) * INDIM + 1];
    tour[(size_t)n * SEQ + step] = (float)pos;
  }
}

// ---------- init / convert / finalize ----------
__global__ void init_state(const float* __restrict__ h0, const float* __restrict__ c0,
                           const float* __restrict__ bos,
                           float* __restrict__ h, float* __restrict__ c,
                           float* __restrict__ prev, float* __restrict__ mask,
                           float* __restrict__ ll)
{
  int idx = blockIdx.x * 256 + threadIdx.x;   // grid covers NB*HID
  h[idx] = h0[idx];
  c[idx] = c0[idx];
  if (idx < NB * SEQ) mask[idx] = 0.0f;
  if (idx < NB * 2)   prev[idx] = bos[idx & 1];
  if (idx < NB)       ll[idx]   = 0.0f;
}

// straight bf16 convert (used for W_hh: its [4H,H] row-major layout IS col-major-by-n B)
__global__ void cvt_bf(const float* __restrict__ in, unsigned short* __restrict__ out, int n)
{
  int idx = blockIdx.x * 256 + threadIdx.x;
  if (idx < n) out[idx] = f2bf(in[idx]);
}

// transpose + convert: in is [K,N] row-major (K=N=HID), out is col-major [N][K]
__global__ void cvt_T(const float* __restrict__ in, unsigned short* __restrict__ out)
{
  int idx = blockIdx.x * 256 + threadIdx.x;   // HID*HID
  int n = idx / HID;
  int k = idx - n * HID;
  out[idx] = f2bf(in[(size_t)k * HID + n]);
}

__global__ void copy_ll(const float* __restrict__ ll, float* __restrict__ out)
{
  int idx = blockIdx.x * 256 + threadIdx.x;
  if (idx < NB) out[idx] = ll[idx];
}

// ---------- host ----------
extern "C" void kernel_launch(void* const* d_in, const int* in_sizes, int n_in,
                              void* d_out, int out_size, void* d_ws, size_t ws_size,
                              hipStream_t stream)
{
  const float* x     = (const float*)d_in[0];
  const float* r     = (const float*)d_in[1];
  const float* h0    = (const float*)d_in[2];
  const float* c0    = (const float*)d_in[3];
  const float* W_ih  = (const float*)d_in[4];
  const float* W_hh  = (const float*)d_in[5];
  const float* b_ih  = (const float*)d_in[6];
  const float* b_hh  = (const float*)d_in[7];
  const float* w_ref = (const float*)d_in[8];
  const float* w_q   = (const float*)d_in[9];
  const float* vv    = (const float*)d_in[10];
  const float* bos   = (const float*)d_in[11];

  char* ws = (char*)d_ws;
  size_t off = 0;
  auto carve = [&](size_t bytes) -> void* {
    void* p = ws + off;
    off = (off + bytes + 255) & ~(size_t)255;
    return p;
  };

  unsigned short* rW   = (unsigned short*)carve((size_t)NB * SEQ * HID * 2);  // 67 MB, L2-resident
  unsigned short* Whhb = (unsigned short*)carve((size_t)HID * G4 * 2);        // col-major == native
  unsigned short* Wqb  = (unsigned short*)carve((size_t)HID * HID * 2);       // transposed
  unsigned short* Wrb  = (unsigned short*)carve((size_t)HID * HID * 2);       // transposed
  float* h     = (float*)carve((size_t)NB * HID * 4);
  float* c     = (float*)carve((size_t)NB * HID * 4);
  float* gates = (float*)carve((size_t)NB * G4 * 4);
  float* q     = (float*)carve((size_t)NB * HID * 4);
  float* prev  = (float*)carve((size_t)NB * 2 * 4);
  float* mask  = (float*)carve((size_t)NB * SEQ * 4);
  float* ll    = (float*)carve((size_t)NB * 4);

  float* tour = (float*)d_out;                 // [NB, SEQ]
  float* llo  = (float*)d_out + NB * SEQ;      // [NB]

  // setup
  init_state<<<(NB * HID) / 256, 256, 0, stream>>>(h0, c0, bos, h, c, prev, mask, ll);
  cvt_bf<<<(HID * G4) / 256, 256, 0, stream>>>(W_hh, Whhb, HID * G4);  // no transpose needed
  cvt_T<<<(HID * HID) / 256, 256, 0, stream>>>(w_q, Wqb);
  cvt_T<<<(HID * HID) / 256, 256, 0, stream>>>(w_ref, Wrb);

  // loop-invariant encoder projection rW = r @ w_ref  (bf16 output, stays in L2)
  {
    dim3 grd(HID / BN, (NB * SEQ) / BM);
    gemm_k<1><<<grd, 256, 0, stream>>>(r, Wrb, nullptr, rW, NB * SEQ, HID, HID,
                                       nullptr, nullptr, nullptr, nullptr);
  }

  dim3 g_gates(G4 / BN, NB / BM);
  dim3 g_q(HID / BN, NB / BM);

  for (int step = 0; step < SEQ; ++step) {
    gemm_k<2><<<g_gates, 256, 0, stream>>>(h, Whhb, gates, nullptr, NB, G4, HID,
                                           b_ih, b_hh, W_ih, prev);
    lstm_pointwise<<<(NB * HID) / 256, 256, 0, stream>>>(gates, c, h);
    gemm_k<0><<<g_q, 256, 0, stream>>>(h, Wqb, q, nullptr, NB, HID, HID,
                                       nullptr, nullptr, nullptr, nullptr);
    attention_step<<<NB, 256, 0, stream>>>(rW, q, vv, x, mask, ll, prev, tour, step);
  }

  copy_ll<<<2, 256, 0, stream>>>(ll, llo);
}
